// PhysicsLoss_59648505807536
// MI455X (gfx1250) — compile-verified, matrix operand order checked by reading, not run on hardware
//
#include <hip/hip_runtime.h>
#include <hip/hip_bf16.h>
#include <math.h>

typedef __attribute__((ext_vector_type(16))) __bf16 v16bf;
typedef __attribute__((ext_vector_type(8)))  float  v8f;
typedef __attribute__((ext_vector_type(4)))  int    v4i;
typedef __bf16 bf16;

#define LL     101
#define NCH    64
#define BB     64
#define NC_    8192
#define NF_    101
#define NTOT   (NC_ + NF_)
#define NPAD   8320                 /* 65 * 128 */
#define KPROJ  (NCH * LL)           /* 6464 = 101*64 */
#define BETA_  6.0f
#define NU_    0.3f
#define PREF_  (1.0f / (1.0f - NU_ * NU_))
#define CDIV(a,b) (((a)+(b)-1)/(b))

#if defined(__HIP_DEVICE_COMPILE__) && __has_builtin(__builtin_amdgcn_global_load_async_to_lds_b128)
#define USE_ASYNC_LDS 1
#else
#define USE_ASYNC_LDS 0
#endif

#define BKK 64                      /* k-step per LDS stage (two WMMA k-chunks) */
#define LDA 72                      /* padded LDS row stride (elements) */

// ---------------------------------------------------------------------------
// Generic WMMA GEMM:  C(MxN) = act( A(MxK) @ W(KxN) [+bias(cols), rows<blim]
//                                   [+Res(MxN)] )
// A, W, Res, C stored bf16 row-major; accumulate f32 via v_wmma_f32_16x16x32_bf16
// Tiling: BM=128, BN=128, BK=64; 256 threads = 8 waves (4 x 2); each wave owns
// a 32x64 subtile = 2x4 WMMA 16x16 tiles. A tile staged with
// global_load_async_to_lds_b128 when available (interior blocks).
// ---------------------------------------------------------------------------
__global__ __launch_bounds__(256) void gemm_bf16_wmma(
    const bf16* __restrict__ A, const bf16* __restrict__ W,
    const float* __restrict__ bias, const bf16* __restrict__ Res,
    bf16* __restrict__ C, int M, int N, int K, int blim, int act)
{
  __shared__ bf16 As[128 * LDA];   // A tile, row-major [row][k]
  __shared__ bf16 Ws[128 * LDA];   // W tile transposed: [n][k]

  const int tid   = threadIdx.x;
  const int lane  = tid & 31;
  const int wid   = tid >> 5;
  const int waveM = wid >> 1;      // 0..3
  const int waveN = wid & 1;       // 0..1
  const int lhalf = lane >> 4;     // 0/1 : which K half-pattern this lane holds
  const int r16   = lane & 15;
  const int rowBase = blockIdx.y * 128;
  const int colBase = blockIdx.x * 128;
  const bool fullM = (rowBase + 128 <= M);
  const bool fullN = (colBase + 128 <= N);

  v8f acc[2][4];
  const v8f vz = {0.f,0.f,0.f,0.f,0.f,0.f,0.f,0.f};
  for (int i = 0; i < 2; ++i)
    for (int j = 0; j < 4; ++j) acc[i][j] = vz;

  for (int k0 = 0; k0 < K; k0 += BKK) {
    // ---- stage A tile (128 x 64) ----
#if USE_ASYNC_LDS
    if (fullM) {
      // async DMA path: 16B per lane, 4 passes, no VGPR round-trip
      const int row = tid >> 3;            // 0..31
      const int kc  = (tid & 7) << 3;      // 0..56 (16B chunks)
      for (int i = 0; i < 4; ++i) {
        int r = row + 32 * i;
        const bf16* gp = A + (size_t)(rowBase + r) * K + k0 + kc;
        bf16* lp = &As[r * LDA + kc];
        __builtin_amdgcn_global_load_async_to_lds_b128(
            (__attribute__((address_space(1))) v4i*)gp,
            (__attribute__((address_space(3))) v4i*)lp, 0, 0);
      }
    } else
#endif
    {
      const int ar = tid >> 4;             // 0..15
      const int ak = (tid & 15) << 2;      // 0..60
      if (fullM) {
        for (int i = 0; i < 8; ++i) {
          int row = ar + 16 * i;
          uint2 v = *(const uint2*)(A + (size_t)(rowBase + row) * K + k0 + ak);
          *(uint2*)(&As[row * LDA + ak]) = v;
        }
      } else {
        for (int i = 0; i < 8; ++i) {
          int row = ar + 16 * i;
          int grow = rowBase + row;
          uint2 v = make_uint2(0u, 0u);
          if (grow < M) v = *(const uint2*)(A + (size_t)grow * K + k0 + ak);
          *(uint2*)(&As[row * LDA + ak]) = v;
        }
      }
    }
    // ---- stage W tile transposed (Ws[n][k], 64 k-rows x 128 n) ----
    {
      const int wk  = tid >> 5;            // 0..7
      const int wn0 = (tid & 31) << 2;     // 0..124
      if (fullN) {
        for (int i = 0; i < 8; ++i) {
          int k = wk + 8 * i;
          uint2 v = *(const uint2*)(W + (size_t)(k0 + k) * N + colBase + wn0);
          bf16 tmp[4];
          *(uint2*)tmp = v;
          Ws[(wn0 + 0) * LDA + k] = tmp[0];
          Ws[(wn0 + 1) * LDA + k] = tmp[1];
          Ws[(wn0 + 2) * LDA + k] = tmp[2];
          Ws[(wn0 + 3) * LDA + k] = tmp[3];
        }
      } else {
        for (int i = 0; i < 8; ++i) {
          int k = wk + 8 * i;
          uint2 v = make_uint2(0u, 0u);
          if (colBase + wn0 < N) v = *(const uint2*)(W + (size_t)(k0 + k) * N + colBase + wn0);
          bf16 tmp[4];
          *(uint2*)tmp = v;
          Ws[(wn0 + 0) * LDA + k] = tmp[0];
          Ws[(wn0 + 1) * LDA + k] = tmp[1];
          Ws[(wn0 + 2) * LDA + k] = tmp[2];
          Ws[(wn0 + 3) * LDA + k] = tmp[3];
        }
      }
    }
#if USE_ASYNC_LDS
    if (fullM) {
#if __has_builtin(__builtin_amdgcn_s_wait_asynccnt)
      __builtin_amdgcn_s_wait_asynccnt(0);
#else
      asm volatile("s_wait_asynccnt 0" ::: "memory");
#endif
    }
#endif
    __syncthreads();

    // prefetch next k-step tiles into cache (global_prefetch_b8)
    if (k0 + BKK < K) {
      if (fullM) __builtin_prefetch(A + (size_t)(rowBase + (tid >> 3)) * K + k0 + BKK + ((tid & 7) << 3), 0, 1);
      if (fullN) __builtin_prefetch(W + (size_t)(k0 + BKK + (tid >> 5)) * N + colBase + ((tid & 31) << 2), 0, 1);
    }

    // ---- two 32-wide WMMA k-chunks per stage ----
    for (int kk = 0; kk < 2; ++kk) {
      const int ko = kk << 5;
      v16bf afrag[2], bfrag[4];
      for (int mt = 0; mt < 2; ++mt) {
        const bf16* ap = &As[(waveM * 32 + mt * 16 + r16) * LDA + ko + 8 * lhalf];
        ((uint4*)&afrag[mt])[0] = *(const uint4*)ap;
        ((uint4*)&afrag[mt])[1] = *(const uint4*)(ap + 16);
      }
      for (int nt = 0; nt < 4; ++nt) {
        const bf16* bp = &Ws[(waveN * 64 + nt * 16 + r16) * LDA + ko + 8 * lhalf];
        ((uint4*)&bfrag[nt])[0] = *(const uint4*)bp;
        ((uint4*)&bfrag[nt])[1] = *(const uint4*)(bp + 16);
      }
      for (int mt = 0; mt < 2; ++mt)
        for (int nt = 0; nt < 4; ++nt)
          acc[mt][nt] = __builtin_amdgcn_wmma_f32_16x16x32_bf16(
              false, afrag[mt], false, bfrag[nt], (short)0, acc[mt][nt], false, false);
    }
    __syncthreads();
  }

  // ---- epilogue ----
  for (int mt = 0; mt < 2; ++mt)
    for (int nt = 0; nt < 4; ++nt)
      for (int j = 0; j < 8; ++j) {
        int row = rowBase + waveM * 32 + mt * 16 + j + 8 * lhalf;
        int col = colBase + waveN * 64 + nt * 16 + r16;
        if (row < M && col < N) {
          float v = acc[mt][nt][j];
          if (bias && row < blim) v += bias[col];
          if (Res) v += (float)Res[(size_t)row * N + col];
          if (act) v = tanhf(v);
          C[(size_t)row * N + col] = (bf16)v;
        }
      }
}

// ---------------------------------------------------------------------------
// small utility / elementwise kernels
// ---------------------------------------------------------------------------
__global__ void cvt_f32_bf16(const float* __restrict__ s, bf16* __restrict__ d, int n) {
  int i = blockIdx.x * 256 + threadIdx.x;
  if (i < n) d[i] = (bf16)s[i];
}

__global__ void build_x0(const float* __restrict__ f, bf16* __restrict__ x0) {
  int idx = blockIdx.x * 256 + threadIdx.x;
  if (idx >= BB * LL) return;
  int b = idx / LL, l = idx % LL;
  x0[(b * 3 + 0) * LL + l] = (bf16)f[b * 2 * LL + l];
  x0[(b * 3 + 1) * LL + l] = (bf16)f[b * 2 * LL + LL + l];
  x0[(b * 3 + 2) * LL + l] = (bf16)(l * (1.0f / 100.0f));
}

__global__ void conv1d_tanh(const bf16* __restrict__ X, const float* __restrict__ Wt,
                            const float* __restrict__ Bs, bf16* __restrict__ Y, int CI) {
  int idx = blockIdx.x * 256 + threadIdx.x;
  if (idx >= BB * NCH * LL) return;
  int l  = idx % LL;
  int co = (idx / LL) % NCH;
  int b  = idx / (LL * NCH);
  float acc = Bs[co];
  for (int ci = 0; ci < CI; ++ci) {
    const bf16* xr = X + (size_t)(b * CI + ci) * LL;
    const float* wr = Wt + (size_t)(co * CI + ci) * 5;
    for (int t = 0; t < 5; ++t) {
      int li = l + t - 2;
      if (li >= 0 && li < LL) acc += wr[t] * (float)xr[li];
    }
  }
  Y[idx] = (bf16)tanhf(acc);
}

__global__ void transpose_bout(const bf16* __restrict__ bout, bf16* __restrict__ boutT) {
  int idx = blockIdx.x * 256 + threadIdx.x;
  if (idx >= 256 * 64) return;
  int k = idx / 64, n = idx % 64;
  boutT[idx] = bout[n * 256 + k];   // boutT[k][n]
}

// lin_in + tanh with fwd-over-fwd seeds -> 6 channel blocks of A
__global__ void trunk_init(const float* __restrict__ xy_i, const float* __restrict__ xy_f,
                           const float* __restrict__ Win, const float* __restrict__ bin,
                           bf16* __restrict__ A) {
  int idx = blockIdx.x * 256 + threadIdx.x;
  if (idx >= NPAD * 256) return;
  int j = idx & 255, p = idx >> 8;
  const size_t CS = (size_t)NPAD * 256;
  if (p < NTOT) {
    float x, y;
    if (p < NC_) { x = xy_i[2 * p]; y = xy_i[2 * p + 1]; }
    else         { x = xy_f[2 * (p - NC_)]; y = xy_f[2 * (p - NC_) + 1]; }
    float w0 = Win[j], w1 = Win[256 + j];
    float z = w0 * x + w1 * y + bin[j];
    float a = tanhf(z), d = 1.f - a * a;
    A[0 * CS + idx] = (bf16)a;
    A[1 * CS + idx] = (bf16)(d * w0);
    A[2 * CS + idx] = (bf16)(d * w1);
    A[3 * CS + idx] = (bf16)(-2.f * a * d * w0 * w0);
    A[4 * CS + idx] = (bf16)(-2.f * a * d * w0 * w1);
    A[5 * CS + idx] = (bf16)(-2.f * a * d * w1 * w1);
  } else {
    for (int c = 0; c < 6; ++c) A[c * CS + idx] = (bf16)0.f;
  }
}

// tanh with second-order forward-mode chain rule, in place over 6 channels
__global__ void tanh_chain(bf16* __restrict__ H) {
  int idx = blockIdx.x * 256 + threadIdx.x;
  if (idx >= NTOT * 256) return;
  const size_t CS = (size_t)NPAD * 256;
  float z0 = (float)H[0 * CS + idx];
  float z1 = (float)H[1 * CS + idx];
  float z2 = (float)H[2 * CS + idx];
  float z3 = (float)H[3 * CS + idx];
  float z4 = (float)H[4 * CS + idx];
  float z5 = (float)H[5 * CS + idx];
  float a = tanhf(z0), d = 1.f - a * a;
  H[0 * CS + idx] = (bf16)a;
  H[1 * CS + idx] = (bf16)(d * z1);
  H[2 * CS + idx] = (bf16)(d * z2);
  H[3 * CS + idx] = (bf16)(d * (z3 - 2.f * a * z1 * z1));
  H[4 * CS + idx] = (bf16)(d * (z4 - 2.f * a * z1 * z2));
  H[5 * CS + idx] = (bf16)(d * (z5 - 2.f * a * z2 * z2));
}

// PDE coefficient assembly for interior loss: C1/C2 (NC x 256)
__global__ void build_C(const bf16* __restrict__ T, const float* __restrict__ xy,
                        bf16* __restrict__ C1, bf16* __restrict__ C2) {
  int idx = blockIdx.x * 256 + threadIdx.x;
  if (idx >= NC_ * 256) return;
  int j = idx & 255, p = idx >> 8;
  const size_t CS = (size_t)NPAD * 256;
  size_t base = (size_t)p * 256 + j;
  float t0 = (float)T[0 * CS + base], t1 = (float)T[1 * CS + base];
  float t2 = (float)T[2 * CS + base], t3 = (float)T[3 * CS + base];
  float t4 = (float)T[4 * CS + base], t5 = (float)T[5 * CS + base];
  float x = xy[2 * p];
  float e = __expf(-x * (1.f / BETA_));
  float r = 1.f - e, rx = e * (1.f / BETA_), rxx = -e * (1.f / (BETA_ * BETA_));
  float hxx = rxx * t0 + 2.f * rx * t1 + r * t3;  // d2(r*s)/dx2
  float hxy = rx * t2 + r * t4;                   // d2(r*s)/dxdy
  float hyy = r * t5;                             // d2(r*s)/dy2
  float c1, c2;
  if (j < 128) { // contracts with b_u
    c1 = PREF_ * (hxx + 0.5f * (1.f - NU_) * hyy);
    c2 = PREF_ * 0.5f * (1.f + NU_) * hxy;
  } else {       // contracts with b_v
    c1 = PREF_ * 0.5f * (1.f + NU_) * hxy;
    c2 = PREF_ * (hyy + 0.5f * (1.f - NU_) * hxx);
  }
  C1[idx] = (bf16)c1;
  C2[idx] = (bf16)c2;
}

// coefficient assembly for forcing loss: D1/D2 (NF x 256)
__global__ void build_D(const bf16* __restrict__ T, const float* __restrict__ xyf,
                        bf16* __restrict__ D1, bf16* __restrict__ D2) {
  int idx = blockIdx.x * 256 + threadIdx.x;
  if (idx >= NF_ * 256) return;
  int j = idx & 255, p = idx >> 8;
  const size_t CS = (size_t)NPAD * 256;
  size_t base = (size_t)(NC_ + p) * 256 + j;
  float t0 = (float)T[0 * CS + base];
  float t1 = (float)T[1 * CS + base];
  float t2 = (float)T[2 * CS + base];
  float x = xyf[2 * p];
  float e = __expf(-x * (1.f / BETA_));
  float r = 1.f - e, rx = e * (1.f / BETA_);
  float sx = rx * t0 + r * t1;   // d(r*s)/dx
  float sy = r * t2;             // d(r*s)/dy
  float d1, d2;
  if (j < 128) { d1 = PREF_ * sx;        d2 = PREF_ * 0.5f * (1.f - NU_) * sy; }
  else         { d1 = PREF_ * NU_ * sy;  d2 = PREF_ * 0.5f * (1.f - NU_) * sx; }
  D1[idx] = (bf16)d1;
  D2[idx] = (bf16)d2;
}

__global__ void build_gT(const float* __restrict__ f, bf16* __restrict__ gxT,
                         bf16* __restrict__ gyT) {
  int idx = blockIdx.x * 256 + threadIdx.x;
  if (idx >= NF_ * BB) return;
  int p = idx / BB, b = idx % BB;
  gxT[p * BB + b] = (bf16)f[b * 2 * LL + p];
  gyT[p * BB + b] = (bf16)f[b * 2 * LL + LL + p];
}

__global__ void zero_out(float* o) { if (threadIdx.x == 0) o[0] = 0.f; }

// deterministic single-block sum of squares, accumulates scale*sum into out[0]
__global__ void sqsum(const bf16* __restrict__ X, int n, float scale, float* __restrict__ out) {
  __shared__ float red[256];
  int tid = threadIdx.x;
  float s = 0.f;
  for (int i = tid; i < n; i += 256) {
    float v = (float)X[i];
    s += v * v;
  }
  red[tid] = s;
  __syncthreads();
  for (int st = 128; st > 0; st >>= 1) {
    if (tid < st) red[tid] += red[tid + st];
    __syncthreads();
  }
  if (tid == 0) out[0] = out[0] + red[0] * scale;
}

// ---------------------------------------------------------------------------
// host driver
// ---------------------------------------------------------------------------
static inline size_t aln(size_t x) { return (x + 255) & ~(size_t)255; }

extern "C" void kernel_launch(void* const* d_in, const int* in_sizes, int n_in,
                              void* d_out, int out_size, void* d_ws, size_t ws_size,
                              hipStream_t stream) {
  (void)in_sizes; (void)n_in; (void)out_size; (void)ws_size;
  // depth-first insertion-order flatten of setup_inputs()
  const float* bcw[3] = {(const float*)d_in[0], (const float*)d_in[2], (const float*)d_in[4]};
  const float* bcb[3] = {(const float*)d_in[1], (const float*)d_in[3], (const float*)d_in[5]};
  const float* fcproj_w = (const float*)d_in[6];
  const float* fcproj_b = (const float*)d_in[7];
  const float* bres_w1[3] = {(const float*)d_in[8],  (const float*)d_in[12], (const float*)d_in[16]};
  const float* bres_b1[3] = {(const float*)d_in[9],  (const float*)d_in[13], (const float*)d_in[17]};
  const float* bres_w2[3] = {(const float*)d_in[10], (const float*)d_in[14], (const float*)d_in[18]};
  const float* bres_b2[3] = {(const float*)d_in[11], (const float*)d_in[15], (const float*)d_in[19]};
  const float* fcout_w = (const float*)d_in[20];
  const float* fcout_b = (const float*)d_in[21];
  const float* linin_w = (const float*)d_in[22];
  const float* linin_b = (const float*)d_in[23];
  const float* tres_w1[4] = {(const float*)d_in[24], (const float*)d_in[28], (const float*)d_in[32], (const float*)d_in[36]};
  const float* tres_b1[4] = {(const float*)d_in[25], (const float*)d_in[29], (const float*)d_in[33], (const float*)d_in[37]};
  const float* tres_w2[4] = {(const float*)d_in[26], (const float*)d_in[30], (const float*)d_in[34], (const float*)d_in[38]};
  const float* tres_b2[4] = {(const float*)d_in[27], (const float*)d_in[31], (const float*)d_in[35], (const float*)d_in[39]};
  const float* linout_w = (const float*)d_in[40];
  const float* linout_b = (const float*)d_in[41];
  const float* f_batch  = (const float*)d_in[42];
  const float* xy_int   = (const float*)d_in[43];
  const float* xy_f     = (const float*)d_in[44];

  char* ws = (char*)d_ws;
  size_t off = 0;
  auto A8 = [&](size_t elems) -> bf16* {
    bf16* p = (bf16*)(ws + off);
    off = aln(off + elems * sizeof(bf16));
    return p;
  };

  bf16* wproj = A8((size_t)KPROJ * 256);
  bf16 *wb1[3], *wb2[3];
  for (int i = 0; i < 3; ++i) { wb1[i] = A8(65536); wb2[i] = A8(65536); }
  bf16* wbout = A8(65536);
  bf16 *wt1[4], *wt2[4];
  for (int i = 0; i < 4; ++i) { wt1[i] = A8(65536); wt2[i] = A8(65536); }
  bf16* wlout = A8(65536);
  bf16* x0   = A8((size_t)BB * 3 * LL);
  bf16* cva  = A8((size_t)BB * NCH * LL);
  bf16* cvb  = A8((size_t)BB * NCH * LL);
  bf16* Xb   = A8((size_t)BB * 256);
  bf16* Hb   = A8((size_t)BB * 256);
  bf16* bout = A8((size_t)BB * 256);
  bf16* boutT = A8(256 * 64);
  bf16* Abuf = A8((size_t)6 * NPAD * 256);
  bf16* Hbuf = A8((size_t)6 * NPAD * 256);
  bf16* C1 = A8((size_t)NC_ * 256);
  bf16* C2 = A8((size_t)NC_ * 256);
  bf16* D1 = A8((size_t)NF_ * 256);
  bf16* D2 = A8((size_t)NF_ * 256);
  bf16* gxT = A8((size_t)NF_ * BB);
  bf16* gyT = A8((size_t)NF_ * BB);
  bf16* LIU = A8((size_t)NC_ * 64);
  bf16* LIV = A8((size_t)NC_ * 64);
  bf16* LFU = A8((size_t)NF_ * 64);
  bf16* LFV = A8((size_t)NF_ * 64);

  auto cvt = [&](const float* s, bf16* d, int n) {
    cvt_f32_bf16<<<CDIV(n, 256), 256, 0, stream>>>(s, d, n);
  };
  auto gemm = [&](const bf16* A, const bf16* W, const float* bias, const bf16* Res,
                  bf16* Cp, int M, int N, int K, int blim, int act) {
    dim3 g(CDIV(N, 128), CDIV(M, 128));
    gemm_bf16_wmma<<<g, 256, 0, stream>>>(A, W, bias, Res, Cp, M, N, K, blim, act);
  };

  // ---- weight conversion to bf16 ----
  cvt(fcproj_w, wproj, KPROJ * 256);
  for (int i = 0; i < 3; ++i) { cvt(bres_w1[i], wb1[i], 65536); cvt(bres_w2[i], wb2[i], 65536); }
  cvt(fcout_w, wbout, 65536);
  for (int i = 0; i < 4; ++i) { cvt(tres_w1[i], wt1[i], 65536); cvt(tres_w2[i], wt2[i], 65536); }
  cvt(linout_w, wlout, 65536);

  // ---- branch network ----
  build_x0<<<CDIV(BB * LL, 256), 256, 0, stream>>>(f_batch, x0);
  conv1d_tanh<<<CDIV(BB * NCH * LL, 256), 256, 0, stream>>>(x0, bcw[0], bcb[0], cva, 3);
  conv1d_tanh<<<CDIV(BB * NCH * LL, 256), 256, 0, stream>>>(cva, bcw[1], bcb[1], cvb, NCH);
  conv1d_tanh<<<CDIV(BB * NCH * LL, 256), 256, 0, stream>>>(cvb, bcw[2], bcb[2], cva, NCH);
  gemm(cva, wproj, fcproj_b, nullptr, Xb, BB, 256, KPROJ, 1 << 30, 1);
  for (int i = 0; i < 3; ++i) {
    gemm(Xb, wb1[i], bres_b1[i], nullptr, Hb, BB, 256, 256, 1 << 30, 1);
    gemm(Hb, wb2[i], bres_b2[i], Xb, Xb, BB, 256, 256, 1 << 30, 0);
  }
  gemm(Xb, wbout, fcout_b, nullptr, bout, BB, 256, 256, 1 << 30, 0);
  transpose_bout<<<CDIV(256 * 64, 256), 256, 0, stream>>>(bout, boutT);

  // ---- trunk forward-over-forward (6 channel blocks of NPAD rows) ----
  trunk_init<<<CDIV(NPAD * 256, 256), 256, 0, stream>>>(xy_int, xy_f, linin_w, linin_b, Abuf);
  const int MT = 6 * NPAD;
  for (int r = 0; r < 4; ++r) {
    gemm(Abuf, wt1[r], tres_b1[r], nullptr, Hbuf, MT, 256, 256, NPAD, 0); // bias: primal rows only
    tanh_chain<<<CDIV(NTOT * 256, 256), 256, 0, stream>>>(Hbuf);
    gemm(Hbuf, wt2[r], tres_b2[r], Abuf, Abuf, MT, 256, 256, NPAD, 0);    // +residual, in place
  }
  gemm(Abuf, wlout, linout_b, nullptr, Hbuf, MT, 256, 256, NPAD, 0);       // T := Hbuf

  // ---- contraction GEMMs ----
  build_C<<<CDIV(NC_ * 256, 256), 256, 0, stream>>>(Hbuf, xy_int, C1, C2);
  build_D<<<CDIV(NF_ * 256, 256), 256, 0, stream>>>(Hbuf, xy_f, D1, D2);
  build_gT<<<CDIV(NF_ * BB, 256), 256, 0, stream>>>(f_batch, gxT, gyT);
  gemm(C1, boutT, nullptr, nullptr, LIU, NC_, 64, 256, 0, 0);
  gemm(C2, boutT, nullptr, nullptr, LIV, NC_, 64, 256, 0, 0);
  gemm(D1, boutT, nullptr, gxT, LFU, NF_, 64, 256, 0, 0);
  gemm(D2, boutT, nullptr, gyT, LFV, NF_, 64, 256, 0, 0);

  // ---- deterministic reduction to scalar loss ----
  float* out = (float*)d_out;
  zero_out<<<1, 1, 0, stream>>>(out);
  const float si = 1.0f / ((float)NC_ * (float)BB);
  const float sf = 1.0f / ((float)BB * (float)NF_);
  sqsum<<<1, 256, 0, stream>>>(LIU, NC_ * 64, si, out);
  sqsum<<<1, 256, 0, stream>>>(LIV, NC_ * 64, si, out);
  sqsum<<<1, 256, 0, stream>>>(LFU, NF_ * 64, sf, out);
  sqsum<<<1, 256, 0, stream>>>(LFV, NF_ * 64, sf, out);
}
